// CapsNet_37984690766357
// MI455X (gfx1250) — matrix operand order repaired and checked by reading, not compiled, not run on hardware
//
#include <hip/hip_runtime.h>

// ---------------------------------------------------------------------------
// CapsNet forward for MI455X (gfx1250, wave32, WMMA).
//   GEMM1: [512*400 x 96pad] x [96 x 256]      (conv1, K=81 padded to 3*32)
//   GEMM2: [512*8*36 x 2592] x [2592 x 32]     (primary caps, K = 81*32 exact)
// Routing is degenerate in the reference (b-logits never updated), so the tail
// collapses to a mean over capsules + squash + 10-way linear + softmax.
//
//  * gemm2: 32x32 per-wave tile (4 WMMA/chunk), B chunk staged in LDS,
//    double-buffered with global_load_async_to_lds_b128 + s_wait_asynccnt.
//  * both gemms: C fragments transposed through LDS so global stores are
//    contiguous b128 runs instead of 2B/4B scatters.
// ---------------------------------------------------------------------------

typedef __attribute__((ext_vector_type(4)))  int    i4;
typedef __attribute__((ext_vector_type(16))) __bf16 v16bf;
typedef __attribute__((ext_vector_type(8)))  float  v8f;

union ABu { i4 q[2]; v16bf v; };
union H8  { __bf16 h[8]; i4 q; };

#define BATCH   512
#define N1      (BATCH * 400)        // 204800 conv1 output positions
#define N2      (BATCH * 8 * 36)     // 147456 primary-caps output positions
#define KC2     81                   // K chunks of 32 for gemm2 (K = 2592)

// --------- async global->LDS (CDNA5), guarded; fallback = load + ds_store --
// Signature confirmed by compiler diagnostics:
//   (addrspace(1) i4* gsrc, addrspace(3) i4* ldst, int offset, int cpol)
#if defined(__HIP_DEVICE_COMPILE__) && __has_builtin(__builtin_amdgcn_global_load_async_to_lds_b128)
  #define HAVE_ASYNC 1
  typedef __attribute__((address_space(1))) i4 gi4;
  typedef __attribute__((address_space(3))) i4 li4;
  #define GLOBAL_TO_LDS16(g, l) \
      __builtin_amdgcn_global_load_async_to_lds_b128((gi4*)(g), (li4*)(l), 0, 0)
  #if __has_builtin(__builtin_amdgcn_s_wait_asynccnt)
    #define WAIT_ASYNC() __builtin_amdgcn_s_wait_asynccnt(0)
  #else
    #define WAIT_ASYNC() asm volatile("s_wait_asynccnt 0x0" ::: "memory")
  #endif
#else
  #define HAVE_ASYNC 0
#endif

// ---------------- prep: im2col for conv1 (pad K 81 -> 96), b128 stores -----
__global__ void k_im2col1(const float* __restrict__ x, __bf16* __restrict__ P) {
    int idx = blockIdx.x * 256 + threadIdx.x;     // < N1 * 12
    int grp = idx % 12, n = idx / 12;
    int b = n / 400, rem = n % 400, y = rem / 20, xx = rem % 20;
    int c0 = grp * 8;
    H8 v8;
    for (int j = 0; j < 8; ++j) {
        int c = c0 + j;
        float v = 0.0f;
        if (c < 81) {
            int kh = c / 9, kw = c % 9;
            v = x[b * 784 + (y + kh) * 28 + (xx + kw)];
        }
        v8.h[j] = (__bf16)v;
    }
    *(i4*)(P + n * 96 + c0) = v8.q;
}

// ---------------- prep: conv1 weights -> WMMA B-operand register image -----
// layout: [kc(3)][g(16)][lane(32)][j(16)]
__global__ void k_pack_wb1(const float* __restrict__ w, __bf16* __restrict__ WB1) {
    int idx = blockIdx.x * 256 + threadIdx.x;     // < 3*16*32*16 = 24576
    int j = idx & 15, lane = (idx >> 4) & 31, g = (idx >> 9) & 15, kc = idx >> 13;
    int k = kc * 32 + ((lane & 16) ? 8 : 0) + (j & 7) + ((j & 8) ? 16 : 0);
    int m = g * 16 + (lane & 15);
    float v = (k < 81) ? w[m * 81 + k] : 0.0f;
    WB1[idx] = (__bf16)v;
}

// ---------------- prep: prim weights -> WMMA B-operand register image ------
// K order: k = (kh*9+kw)*32 + i  (matches contiguous channel-last A2 gathers)
__global__ void k_pack_wb2(const float* __restrict__ w, __bf16* __restrict__ WB2) {
    int idx = blockIdx.x * 256 + threadIdx.x;     // < 81*2*32*16 = 82944
    int j = idx & 15, lane = (idx >> 4) & 31, g = (idx >> 9) & 1, kc = idx >> 10;
    int k = kc * 32 + ((lane & 16) ? 8 : 0) + (j & 7) + ((j & 8) ? 16 : 0);
    int i = k & 31, p = k >> 5, kh = p / 9, kw = p % 9;
    int o = g * 16 + (lane & 15);
    WB2[idx] = (__bf16)w[(o * 32 + i) * 81 + kh * 9 + kw];
}

// ---------------- prep: tail constants -------------------------------------
__global__ void k_pack_tail(const float* __restrict__ out_w, const float* __restrict__ dig_Wb,
                            float* __restrict__ sumW, float* __restrict__ sumWb) {
    int t = threadIdx.x;
    if (t < 160) {
        int o = t / 16, e = t % 16;
        float s = 0.0f;
        for (int i = 0; i < 10; ++i) s += out_w[(o * 10 + i) * 16 + e];
        sumW[t] = s;
    } else if (t < 176) {
        int e = t - 160;
        float s = 0.0f;
        for (int n = 0; n < 1152; ++n) s += dig_Wb[n * 16 + e];
        sumWb[e] = s;
    }
}

// ---------------- GEMM1: conv1 as WMMA GEMM + bias + relu ------------------
// grid 12800 x 128 (4 waves). WG = 16 positions x 256 channels; wave wv owns
// 64 channels. C goes through LDS in A2-order so global stores are 64B runs.
__global__ __launch_bounds__(128) void k_gemm1(const __bf16* __restrict__ P,
                                               const __bf16* __restrict__ WB1,
                                               const float* __restrict__ bias,
                                               __bf16* __restrict__ A2) {
    __shared__ __bf16 Cs[16 * 256];               // [pos][d*32 + i], 8 KB
    int t = threadIdx.x, lane = t & 31, wv = t >> 5;
    int hi = lane >> 4, m = lane & 15;
    int rowbase = blockIdx.x * 16;
    int row = rowbase + m;
    const __bf16* Ar = P + row * 96 + hi * 8;

    v8f zero = {0.f,0.f,0.f,0.f,0.f,0.f,0.f,0.f};
    v8f acc[4];
    for (int g = 0; g < 4; ++g) acc[g] = zero;

    for (int kc = 0; kc < 3; ++kc) {
        ABu a;
        a.q[0] = *(const i4*)(Ar + kc * 32);
        a.q[1] = *(const i4*)(Ar + kc * 32 + 16);
        for (int g = 0; g < 4; ++g) {
            const __bf16* bp = WB1 + ((kc * 16 + (wv * 4 + g)) * 32 + lane) * 16;
            ABu bm;
            bm.q[0] = *(const i4*)(bp);
            bm.q[1] = *(const i4*)(bp + 8);
            acc[g] = __builtin_amdgcn_wmma_f32_16x16x32_bf16(
                false, a.v, false, bm.v, (short)0, acc[g], false, false);
        }
    }
    // scatter into LDS in A2-order (cheap 2B LDS writes, not global)
    for (int g = 0; g < 4; ++g) {
        int ch = wv * 64 + g * 16 + m;
        float bs = bias[ch];
        int dst = (ch & 7) * 32 + (ch >> 3);
        for (int r = 0; r < 8; ++r) {
            int pl = hi * 8 + r;
            float v = acc[g][r] + bs;
            v = v > 0.f ? v : 0.f;
            Cs[pl * 256 + dst] = (__bf16)v;
        }
    }
    __syncthreads();
    // 128 threads write 128 contiguous 64B runs
    int pl = t >> 3, dd = t & 7;
    const __bf16* src = &Cs[pl * 256 + dd * 32];
    int pos = rowbase + pl;
    int bb = pos / 400, rem = pos % 400, y = rem / 20, xx = rem % 20;
    __bf16* dst = A2 + (((bb * 8 + dd) * 20 + y) * 20 + xx) * 32;
    ((i4*)dst)[0] = ((const i4*)src)[0];
    ((i4*)dst)[1] = ((const i4*)src)[1];
    ((i4*)dst)[2] = ((const i4*)src)[2];
    ((i4*)dst)[3] = ((const i4*)src)[3];
}

// ---------------- GEMM2: primary caps as WMMA GEMM + bias + relu -----------
// grid 1152 x 128. Wave = "squad": 1 batch image b, 4 spatial positions
// s0..s0+3, all 8 depths, all 32 out-channels => two 16-row A tiles with row
// mapping M -> (d = M>>1, s = s0 + tile*2 + (M&1)). 4 WMMA per K chunk.
// B chunk (2KB) staged in LDS, double buffered (async when available).
// C goes through LDS so U stores are contiguous 32B runs.
__global__ __launch_bounds__(128) void k_gemm2(const __bf16* __restrict__ A2,
                                               const __bf16* __restrict__ WB2,
                                               const float* __restrict__ prim_b,
                                               float* __restrict__ U) {
    __shared__ __bf16 Bb[2][1024];                // 4 KB, double-buffered B chunk
    __shared__ float  Cs[4 * 4 * 32 * 8];         // [wv][sl][o][d], 16 KB
    int t = threadIdx.x, lane = t & 31, wv = t >> 5;
    int hi = lane >> 4, m = lane & 15;

    int S = blockIdx.x * 4 + wv;                  // squad id < 4608
    int b = S / 9, s0 = (S % 9) * 4;

    int d   = m >> 1;
    int sa0 = s0 + (m & 1);                       // tile 0 spatial index
    int sa1 = sa0 + 2;                            // tile 1 spatial index
    int yo0 = sa0 / 6, xo0 = sa0 % 6;
    int yo1 = sa1 / 6, xo1 = sa1 % 6;
    const __bf16* Ac0 = A2 + (((b * 8 + d) * 20 + 2 * yo0) * 20 + 2 * xo0) * 32 + hi * 8;
    const __bf16* Ac1 = A2 + (((b * 8 + d) * 20 + 2 * yo1) * 20 + 2 * xo1) * 32 + hi * 8;

    // preload B chunk 0
#if HAVE_ASYNC
    GLOBAL_TO_LDS16(WB2 + t * 8, &Bb[0][t * 8]);
#else
    { i4 p0 = *(const i4*)(WB2 + t * 8); *(i4*)(&Bb[0][t * 8]) = p0; }
#endif

    v8f zero = {0.f,0.f,0.f,0.f,0.f,0.f,0.f,0.f};
    v8f acc00 = zero, acc01 = zero, acc10 = zero, acc11 = zero;

    for (int kc = 0; kc < KC2; ++kc) {
        int cur = kc & 1, nxt = cur ^ 1;
#if HAVE_ASYNC
        WAIT_ASYNC();                             // our part of buf[cur] landed
        __syncthreads();                          // everyone's landed; reads of buf[nxt] done
        if (kc + 1 < KC2)
            GLOBAL_TO_LDS16(WB2 + (kc + 1) * 1024 + t * 8, &Bb[nxt][t * 8]);
#else
        i4 pf;
        if (kc + 1 < KC2) pf = *(const i4*)(WB2 + (kc + 1) * 1024 + t * 8);
        __syncthreads();
        if (kc + 1 < KC2) *(i4*)(&Bb[nxt][t * 8]) = pf;
#endif
        int kh = kc / 9, kw = kc % 9;
        int aoff = (kh * 20 + kw) * 32;
        ABu a0, a1, b0, b1;
        a0.q[0] = *(const i4*)(Ac0 + aoff);
        a0.q[1] = *(const i4*)(Ac0 + aoff + 16);
        a1.q[0] = *(const i4*)(Ac1 + aoff);
        a1.q[1] = *(const i4*)(Ac1 + aoff + 16);
        const __bf16* bp0 = &Bb[cur][lane * 16];
        const __bf16* bp1 = &Bb[cur][(32 + lane) * 16];
        b0.q[0] = *(const i4*)(bp0); b0.q[1] = *(const i4*)(bp0 + 8);
        b1.q[0] = *(const i4*)(bp1); b1.q[1] = *(const i4*)(bp1 + 8);
        acc00 = __builtin_amdgcn_wmma_f32_16x16x32_bf16(false, a0.v, false, b0.v, (short)0, acc00, false, false);
        acc01 = __builtin_amdgcn_wmma_f32_16x16x32_bf16(false, a0.v, false, b1.v, (short)0, acc01, false, false);
        acc10 = __builtin_amdgcn_wmma_f32_16x16x32_bf16(false, a1.v, false, b0.v, (short)0, acc10, false, false);
        acc11 = __builtin_amdgcn_wmma_f32_16x16x32_bf16(false, a1.v, false, b1.v, (short)0, acc11, false, false);
    }

    __syncthreads();                              // Bb done; reuse LDS phase for Cs
    // scatter C into LDS in U-order: Cs[((wv*4 + sl)*32 + o)*8 + d]
    float bs0 = prim_b[m], bs1 = prim_b[16 + m];
    for (int r = 0; r < 8; ++r) {
        int M = hi * 8 + r;
        int dd = M >> 1, si = M & 1;
        float v;
        v = acc00[r] + bs0; v = v > 0.f ? v : 0.f;
        Cs[((wv * 4 + si)     * 32 + m)      * 8 + dd] = v;
        v = acc01[r] + bs1; v = v > 0.f ? v : 0.f;
        Cs[((wv * 4 + si)     * 32 + 16 + m) * 8 + dd] = v;
        v = acc10[r] + bs0; v = v > 0.f ? v : 0.f;
        Cs[((wv * 4 + 2 + si) * 32 + m)      * 8 + dd] = v;
        v = acc11[r] + bs1; v = v > 0.f ? v : 0.f;
        Cs[((wv * 4 + 2 + si) * 32 + 16 + m) * 8 + dd] = v;
    }
    __syncthreads();
    // 512 contiguous 32B runs -> U
    for (int j = 0; j < 4; ++j) {
        int R = t + j * 128;                      // ((wvr*4 + sl)*32 + o)
        int o = R & 31, sl = (R >> 5) & 3, wvr = R >> 7;
        int Sr = blockIdx.x * 4 + wvr;
        int br = Sr / 9, s = (Sr % 9) * 4 + sl;
        int yo = s / 6, xo = s % 6;
        float* dst = U + (br * 1152 + o * 36 + xo * 6 + yo) * 8;
        const float* src = &Cs[R * 8];
        ((i4*)dst)[0] = ((const i4*)src)[0];
        ((i4*)dst)[1] = ((const i4*)src)[1];
    }
}

// ---------------- tail: sv[b,e] = (sum_n sum_d u*W + sum_n Wb)/1152 --------
__global__ __launch_bounds__(256) void k_tail_reduce(const float* __restrict__ U,
                                                     const float* __restrict__ digW,
                                                     const float* __restrict__ sumWb,
                                                     float* __restrict__ sv) {
    __shared__ float red[256 * 16];
    int b = blockIdx.x, t = threadIdx.x;
    float acc[16];
    for (int e = 0; e < 16; ++e) acc[e] = 0.0f;
    for (int n = t; n < 1152; n += 256) {
        const float* up = U + (b * 1152 + n) * 8;
        const float* wp = digW + n * 128;
        for (int dd = 0; dd < 8; ++dd) {
            float ud = up[dd];
            for (int e = 0; e < 16; ++e) acc[e] += ud * wp[dd * 16 + e];
        }
    }
    for (int e = 0; e < 16; ++e) red[t * 16 + e] = acc[e];
    __syncthreads();
    for (int s = 128; s >= 1; s >>= 1) {
        if (t < s)
            for (int e = 0; e < 16; ++e) red[t * 16 + e] += red[(t + s) * 16 + e];
        __syncthreads();
    }
    if (t < 16) sv[b * 16 + t] = (red[t] + sumWb[t]) * (1.0f / 1152.0f);
}

// ---------------- tail: squash + linear + softmax --------------------------
__global__ void k_tail_final(const float* __restrict__ sv, const float* __restrict__ sumW,
                             const float* __restrict__ out_b, float* __restrict__ out) {
    int b = blockIdx.x * 256 + threadIdx.x;
    if (b >= BATCH) return;
    float s[16], l2 = 0.0f, l1 = 0.0f;
    for (int e = 0; e < 16; ++e) {
        float v = sv[b * 16 + e];
        s[e] = v; l2 += v * v; l1 += fabsf(v);
    }
    l2 = sqrtf(l2);
    float scale = (l2 / (1.0f + l2)) / l1;
    float logits[10], mx = -1e30f;
    for (int o = 0; o < 10; ++o) {
        float a = out_b[o];
        for (int e = 0; e < 16; ++e) a += s[e] * scale * sumW[o * 16 + e];
        logits[o] = a; mx = fmaxf(mx, a);
    }
    float den = 0.0f;
    for (int o = 0; o < 10; ++o) { logits[o] = expf(logits[o] - mx); den += logits[o]; }
    for (int o = 0; o < 10; ++o) out[b * 10 + o] = logits[o] / den;
}

// ---------------------------------------------------------------------------
extern "C" void kernel_launch(void* const* d_in, const int* in_sizes, int n_in,
                              void* d_out, int out_size, void* d_ws, size_t ws_size,
                              hipStream_t stream) {
    const float* x       = (const float*)d_in[0];
    const float* conv1_w = (const float*)d_in[1];
    const float* conv1_b = (const float*)d_in[2];
    const float* prim_w  = (const float*)d_in[3];
    const float* prim_b  = (const float*)d_in[4];
    const float* dig_W   = (const float*)d_in[5];
    const float* dig_Wb  = (const float*)d_in[6];
    const float* out_w   = (const float*)d_in[7];
    const float* out_b   = (const float*)d_in[8];

    char* ws = (char*)d_ws;
    // workspace layout (256B aligned), total ~163.4 MB
    const size_t off_P     = 0;                              // 204800*96*2   = 39,321,600
    const size_t off_WB1   = off_P     + 39321600;           // 24576*2       = 49,152
    const size_t off_A2    = off_WB1   + 49152;              // 512*8*400*32*2= 104,857,600
    const size_t off_WB2   = off_A2    + 104857600;          // 82944*2       = 165,888
    const size_t off_U     = off_WB2   + 165888;             // 512*1152*8*4  = 18,874,368
    const size_t off_sv    = off_U     + 18874368;           // 512*16*4      = 32,768
    const size_t off_sumW  = off_sv    + 32768;              // 640
    const size_t off_sumWb = off_sumW  + 768;                // 64

    __bf16* P    = (__bf16*)(ws + off_P);
    __bf16* WB1  = (__bf16*)(ws + off_WB1);
    __bf16* A2   = (__bf16*)(ws + off_A2);
    __bf16* WB2  = (__bf16*)(ws + off_WB2);
    float*  U    = (float*)(ws + off_U);
    float*  sv   = (float*)(ws + off_sv);
    float*  sumW = (float*)(ws + off_sumW);
    float*  sumWb= (float*)(ws + off_sumWb);

    k_im2col1 <<<9600, 256, 0, stream>>>(x, P);
    k_pack_wb1<<<96,   256, 0, stream>>>(conv1_w, WB1);
    k_pack_wb2<<<324,  256, 0, stream>>>(prim_w, WB2);
    k_pack_tail<<<1,   256, 0, stream>>>(out_w, dig_Wb, sumW, sumWb);

    k_gemm1<<<12800, 128, 0, stream>>>(P, WB1, conv1_b, A2);
    k_gemm2<<<1152,  128, 0, stream>>>(A2, WB2, prim_b, U);

    k_tail_reduce<<<512, 256, 0, stream>>>(U, dig_W, sumWb, sv);
    k_tail_final <<<2,   256, 0, stream>>>(sv, sumW, out_b, (float*)d_out);
}
